// BERT_CRF_3865470567130
// MI455X (gfx1250) — matrix operand order repaired and checked by reading, not compiled, not run on hardware
//
#include <hip/hip_runtime.h>
#include <hip/hip_bf16.h>

// ---------------------------------------------------------------------------
// BERT-CRF: emissions GEMM (WMMA f16, HBM-bound) + Viterbi decode (lane/chain)
//   B=64, T=512, H=768, K=4 (BMES)
// ---------------------------------------------------------------------------

#define B_ 64
#define T_ 512
#define H_ 768
#define K_ 4
#define NROWS (B_ * T_)          // 32768 flattened (b,t) rows

typedef __attribute__((ext_vector_type(16))) _Float16 v16h;
typedef __attribute__((ext_vector_type(8)))  float    v8f;

// ---------------------------------------------------------------------------
// Phase 1: emissions[b,t,k] = sentences[b,t,:] . W[k,:] + bias[k]
// One wave computes one 16(row) x 16(col, K padded 4->16) tile, looping H in
// chunks of 32 with v_wmma_f32_16x16x32_f16. 2048 tiles total.
// ---------------------------------------------------------------------------
__global__ __launch_bounds__(256) void bertcrf_emissions_wmma(
    const float* __restrict__ S,     // [NROWS, H]
    const float* __restrict__ W,     // [K, H]
    const float* __restrict__ bias,  // [K]
    float* __restrict__ E)           // [NROWS, K]  (workspace)
{
    const int lane    = threadIdx.x & 31;
    const int wave    = threadIdx.x >> 5;
    const int tile    = blockIdx.x * 8 + wave;   // 16-row tile id
    const int row0    = tile * 16;
    const int hs      = lane >> 4;               // half-wave select
    const int l16     = lane & 15;

    // A fragment: this lane owns matrix row m = l16.
    // K offsets per 32-chunk: {hs*8 + 0..7} -> elems 0..7, {16 + hs*8 + 0..7} -> elems 8..15
    const float* arow = S + (size_t)(row0 + l16) * H_;

    // B fragment: this lane owns matrix column n = l16 (W row n); n>=K_ is zero pad.
    // K offsets per 32-chunk: {hs*8 + 0..7} -> elems 0..7, {16 + hs*8 + 0..7} -> elems 8..15
    const bool   bvalid = (l16 < K_);
    const float* brow   = bvalid ? (W + (size_t)l16 * H_) : W;

    v8f acc = {};
    for (int h0 = 0; h0 < H_; h0 += 32) {
        const float* pa = arow + h0 + hs * 8;
        // prefetch next chunk of this row (emits global_prefetch_b8)
        __builtin_prefetch(pa + 32, 0, 1);

        float4 a0 = *(const float4*)(pa +  0);
        float4 a1 = *(const float4*)(pa +  4);
        float4 a2 = *(const float4*)(pa + 16);
        float4 a3 = *(const float4*)(pa + 20);
        v16h a;
        a[0]=(_Float16)a0.x; a[1]=(_Float16)a0.y; a[2]=(_Float16)a0.z; a[3]=(_Float16)a0.w;
        a[4]=(_Float16)a1.x; a[5]=(_Float16)a1.y; a[6]=(_Float16)a1.z; a[7]=(_Float16)a1.w;
        a[8]=(_Float16)a2.x; a[9]=(_Float16)a2.y; a[10]=(_Float16)a2.z; a[11]=(_Float16)a2.w;
        a[12]=(_Float16)a3.x; a[13]=(_Float16)a3.y; a[14]=(_Float16)a3.z; a[15]=(_Float16)a3.w;

        v16h bf = {};
        if (bvalid) {
            const float* pb = brow + h0 + hs * 8;
            float4 b0 = *(const float4*)(pb +  0);
            float4 b1 = *(const float4*)(pb +  4);
            float4 b2 = *(const float4*)(pb + 16);
            float4 b3 = *(const float4*)(pb + 20);
            bf[0]=(_Float16)b0.x; bf[1]=(_Float16)b0.y; bf[2]=(_Float16)b0.z; bf[3]=(_Float16)b0.w;
            bf[4]=(_Float16)b1.x; bf[5]=(_Float16)b1.y; bf[6]=(_Float16)b1.z; bf[7]=(_Float16)b1.w;
            bf[8]=(_Float16)b2.x; bf[9]=(_Float16)b2.y; bf[10]=(_Float16)b2.z; bf[11]=(_Float16)b2.w;
            bf[12]=(_Float16)b3.x; bf[13]=(_Float16)b3.y; bf[14]=(_Float16)b3.z; bf[15]=(_Float16)b3.w;
        }

        // 8 args: (neg_a, A, neg_b, B, c_mod, C, reuse_a, reuse_b)
        acc = __builtin_amdgcn_wmma_f32_16x16x32_f16(
            false, a, false, bf, (short)0, acc, false, false);
    }

    // C/D layout: lane col n=l16, accumulator slot r -> row hs*8 + r
    if (bvalid) {
        const float bv = bias[l16];
        const int mbase = hs * 8;
#pragma unroll
        for (int r = 0; r < 8; ++r) {
            const int grow = row0 + mbase + r;
            E[(size_t)grow * K_ + l16] = acc[r] + bv;
        }
    }
}

// ---------------------------------------------------------------------------
// Phase 2: Viterbi decode. One lane per batch chain; 2 blocks x 32 lanes.
// Forward pass stores 4x2-bit backpointers per step packed into one byte,
// then the same lane backtracks and writes tags.
// ---------------------------------------------------------------------------
__global__ __launch_bounds__(32) void bertcrf_viterbi(
    const float* __restrict__ E,       // [NROWS, K] emissions
    const float* __restrict__ startT,  // [K]
    const float* __restrict__ endT,    // [K]
    const float* __restrict__ trans,   // [K, K]
    unsigned char* __restrict__ bp,    // [B, T] packed backpointers (ws)
    float* __restrict__ out)           // [B, T] tags as float
{
    const int b = blockIdx.x * 32 + threadIdx.x;
    if (b >= B_) return;

    float tr[K_][K_];
#pragma unroll
    for (int i = 0; i < K_; ++i)
#pragma unroll
        for (int j = 0; j < K_; ++j)
            tr[i][j] = trans[i * K_ + j];

    const float4* em = (const float4*)(E + (size_t)b * T_ * K_);
    unsigned char* mybp = bp + (size_t)b * T_;

    float4 e0 = em[0];
    float score[K_] = { startT[0] + e0.x, startT[1] + e0.y,
                        startT[2] + e0.z, startT[3] + e0.w };

    for (int t = 1; t < T_; ++t) {
        float4 et = em[t];
        const float emit[K_] = { et.x, et.y, et.z, et.w };
        float ns[K_];
        unsigned code = 0;
#pragma unroll
        for (int j = 0; j < K_; ++j) {
            float best = score[0] + tr[0][j];
            int   bi   = 0;
#pragma unroll
            for (int i = 1; i < K_; ++i) {
                const float c = score[i] + tr[i][j];
                if (c > best) { best = c; bi = i; }   // strict > => first-max (jnp.argmax)
            }
            ns[j] = best + emit[j];
            code |= (unsigned)bi << (2 * j);
        }
        mybp[t] = (unsigned char)code;
#pragma unroll
        for (int j = 0; j < K_; ++j) score[j] = ns[j];
    }

    // final tag: argmax(score + end_transitions), first-max tie break
    int   cur  = 0;
    float best = score[0] + endT[0];
#pragma unroll
    for (int j = 1; j < K_; ++j) {
        const float f = score[j] + endT[j];
        if (f > best) { best = f; cur = j; }
    }
    out[(size_t)b * T_ + (T_ - 1)] = (float)cur;

    for (int t = T_ - 2; t >= 0; --t) {
        cur = (mybp[t + 1] >> (2 * cur)) & 3;
        out[(size_t)b * T_ + t] = (float)cur;
    }
}

// ---------------------------------------------------------------------------
// Launch: inputs in setup_inputs() order:
//   [0] sentences [B,T,H] f32   [1] W [K,H] f32   [2] b [K] f32
//   [3] start_transitions [K]   [4] end_transitions [K]   [5] transitions [K,K]
// Workspace: emissions (NROWS*K_ floats = 512KB) + packed bps (B_*T_ bytes = 32KB)
// ---------------------------------------------------------------------------
extern "C" void kernel_launch(void* const* d_in, const int* in_sizes, int n_in,
                              void* d_out, int out_size, void* d_ws, size_t ws_size,
                              hipStream_t stream) {
    const float* S      = (const float*)d_in[0];
    const float* W      = (const float*)d_in[1];
    const float* bias   = (const float*)d_in[2];
    const float* startT = (const float*)d_in[3];
    const float* endT   = (const float*)d_in[4];
    const float* trans  = (const float*)d_in[5];

    float*         E  = (float*)d_ws;                                   // 512 KB
    unsigned char* bp = (unsigned char*)d_ws + (size_t)NROWS * K_ * sizeof(float);

    float* out = (float*)d_out;   // tags as float, [B,T]

    // 2048 tiles of 16 rows, 8 waves (256 threads) per block -> 256 blocks
    bertcrf_emissions_wmma<<<NROWS / (16 * 8), 256, 0, stream>>>(S, W, bias, E);
    // 64 chains, one per lane, 2 wave32 blocks
    bertcrf_viterbi<<<2, 32, 0, stream>>>(E, startT, endT, trans, bp, out);
}
